// GAT_3_38036230373751
// MI455X (gfx1250) — matrix operand order repaired and checked by reading, not compiled, not run on hardware
//
#include <hip/hip_runtime.h>

// ---------------------------------------------------------------------------
// EdgeGAT pipeline for gfx1250 (MI455X): bf16 WMMA GEMMs + atomic edge ops.
// Round 3: ping-pong double-buffered WMMA GEMM (K unrolled by 64, two named
// fragment sets) -> no register-rotation moves, 8 v_wmma per iteration with
// loads for the next half-step overlapping the current WMMAs.
// ---------------------------------------------------------------------------

#define NNODES 10000
#define NEDGES 160000

typedef __bf16 bf16;
typedef __attribute__((ext_vector_type(16))) __bf16 v16bf;
typedef __attribute__((ext_vector_type(8)))  float  v8f;

__device__ __forceinline__ unsigned short f2bf_rne(float f) {
  unsigned u = __float_as_uint(f);
  unsigned r = u + 0x7FFFu + ((u >> 16) & 1u);   // round-to-nearest-even
  return (unsigned short)(r >> 16);
}

__device__ __forceinline__ float leaky02(float x) {
  return x > 0.f ? x : 0.2f * x;
}

__device__ __forceinline__ void atomicMaxF(float* addr, float v) {
  if (v >= 0.f) atomicMax((int*)addr, __float_as_int(v));
  else          atomicMin((unsigned int*)addr, __float_as_uint(v));
}

// A fragment: lanes 0-15 row M=l15, K = k0..k0+15;
// lanes 16-31 K = k0+16..k0+31 (matches B fragment order).
__device__ __forceinline__ v16bf load_afrag(const bf16* __restrict__ Ab,
                                            int mrow, int K, int k0, int hi) {
  v16bf a;
  const bf16* ap = Ab + (size_t)mrow * K + k0 + hi * 16;
#pragma unroll
  for (int j = 0; j < 8; ++j) a[j] = ap[j];
#pragma unroll
  for (int j = 0; j < 8; ++j) a[8 + j] = ap[8 + j];
  return a;
}

// B fragment: column-major B (Bt is [N][K]); lanes 0-15 col N=l15 K=k0..k0+15,
// lanes 16-31 K=k0+16..k0+31 -> 16 contiguous bf16 per lane (2x b128 loads).
__device__ __forceinline__ v16bf load_bfrag(const bf16* __restrict__ Bb,
                                            int ncol, int K, int k0, int hi) {
  v16bf b;
  const bf16* bp = Bb + (size_t)ncol * K + k0 + hi * 16;
#pragma unroll
  for (int j = 0; j < 16; ++j) b[j] = bp[j];
  return b;
}

// ---------------------------------------------------------------------------
// WMMA GEMM: C[M,N] = A[M,K] * B[K,N], A bf16 row-major, B bf16 pre-transposed
// to [N,K]; fp32 accumulate. One wave computes a 16x64 tile (A frag reused
// across 4 N-subtiles). Requires M%16==0, N%64==0, K%64==0.
// ---------------------------------------------------------------------------
__global__ void __launch_bounds__(256) wmma_gemm_bf16(
    const unsigned short* __restrict__ A,
    const unsigned short* __restrict__ Bt,
    float* __restrict__ C, int M, int N, int K) {
  const int lane  = threadIdx.x & 31;
  const int gwave = blockIdx.x * 8 + (threadIdx.x >> 5);
  const int tilesN = N >> 6;
  const int tileM  = gwave / tilesN;
  const int tileN  = gwave % tilesN;
  if (tileM >= (M >> 4)) return;                 // wave-uniform exit

  const bf16* Ab = (const bf16*)A;
  const bf16* Bb = (const bf16*)Bt;
  const int l15 = lane & 15;
  const int hi  = lane >> 4;
  const int mrow  = (tileM << 4) + l15;
  const int ncolb = (tileN << 6) + l15;

  v8f acc[4];
#pragma unroll
  for (int t = 0; t < 4; ++t) acc[t] = (v8f){};

  // prologue: stage k=0 fragments into set A
  v16bf aA = load_afrag(Ab, mrow, K, 0, hi);
  v16bf bA[4];
#pragma unroll
  for (int t = 0; t < 4; ++t) bA[t] = load_bfrag(Bb, ncolb + (t << 4), K, 0, hi);

  // main loop: K unrolled by 64 (all K used are multiples of 64); ping-pong
  // between fragment sets A and B so no register rotation is needed.
  for (int k0 = 0; k0 < K; k0 += 64) {
    // stage k0+32 into set B (always exists: k0 <= K-64)
    v16bf aB = load_afrag(Ab, mrow, K, k0 + 32, hi);
    v16bf bB[4];
#pragma unroll
    for (int t = 0; t < 4; ++t) bB[t] = load_bfrag(Bb, ncolb + (t << 4), K, k0 + 32, hi);

#pragma unroll
    for (int t = 0; t < 4; ++t)
      acc[t] = __builtin_amdgcn_wmma_f32_16x16x32_bf16(
          false, aA, false, bA[t], (short)0, acc[t], false, false);

    if (k0 + 64 < K) {                           // stage k0+64 into set A
      aA = load_afrag(Ab, mrow, K, k0 + 64, hi);
#pragma unroll
      for (int t = 0; t < 4; ++t) bA[t] = load_bfrag(Bb, ncolb + (t << 4), K, k0 + 64, hi);
    }

#pragma unroll
    for (int t = 0; t < 4; ++t)
      acc[t] = __builtin_amdgcn_wmma_f32_16x16x32_bf16(
          false, aB, false, bB[t], (short)0, acc[t], false, false);
  }

  const int rbase = (tileM << 4) + hi * 8;       // C/D: M split per half-wave
#pragma unroll
  for (int t = 0; t < 4; ++t) {
    const int ccol = (tileN << 6) + (t << 4) + l15;
#pragma unroll
    for (int r = 0; r < 8; ++r)
      C[(size_t)(rbase + r) * N + ccol] = acc[t][r];
  }
}

// ---------------------------------------------------------------------------
// Elementwise / graph kernels
// ---------------------------------------------------------------------------
__global__ void deg_init(float* od, float* id, int n) {
  int i = blockIdx.x * blockDim.x + threadIdx.x;
  if (i < n) { od[i] = 1.0f; id[i] = 1.0f; }     // self-loop contribution
}

__global__ void deg_accum(const int* __restrict__ src, const int* __restrict__ dst,
                          float* od, float* id, int e) {
  int i = blockIdx.x * blockDim.x + threadIdx.x;
  if (i < e) { atomicAdd(&od[src[i]], 1.0f); atomicAdd(&id[dst[i]], 1.0f); }
}

__global__ void deg_norm(const float* od, const float* id, float* sn, float* dn, int n) {
  int i = blockIdx.x * blockDim.x + threadIdx.x;
  if (i < n) { sn[i] = rsqrtf(od[i]); dn[i] = rsqrtf(id[i]); }
}

__global__ void scale_to_bf16(const float* __restrict__ X, const float* __restrict__ rowscale,
                              unsigned short* __restrict__ Y, int rows, int cols) {
  int i = blockIdx.x * blockDim.x + threadIdx.x;
  if (i < rows * cols) {
    float s = rowscale ? rowscale[i / cols] : 1.0f;
    Y[i] = f2bf_rne(X[i] * s);
  }
}

__global__ void w_to_bf16_t(const float* __restrict__ W, unsigned short* __restrict__ Wt,
                            int K, int N) {
  int i = blockIdx.x * blockDim.x + threadIdx.x;
  if (i < K * N) { int k = i / N, n = i - k * N; Wt[(size_t)n * K + k] = f2bf_rne(W[i]); }
}

__global__ void scatter_add4(const float* __restrict__ hw, const int* __restrict__ src,
                             const int* __restrict__ dst, float* __restrict__ agg,
                             int e, int fq, int F) {
  int i = blockIdx.x * blockDim.x + threadIdx.x;
  if (i >= e * fq) return;
  int ed = i / fq, c = (i - ed * fq) << 2;
  const float4 v = *(const float4*)(hw + (size_t)src[ed] * F + c);
  float* q = agg + (size_t)dst[ed] * F + c;
  atomicAdd(q + 0, v.x); atomicAdd(q + 1, v.y);
  atomicAdd(q + 2, v.z); atomicAdd(q + 3, v.w);
}

__global__ void post_relu(float* __restrict__ agg, const float* __restrict__ dn,
                          const float* __restrict__ b, int rows, int F) {
  int i = blockIdx.x * blockDim.x + threadIdx.x;
  if (i < rows * F) {
    int r = i / F, f = i - r * F;
    float t = agg[i] * dn[r] + b[f];
    agg[i] = t > 0.f ? t : 0.f;
  }
}

__global__ void bn_stats(const float* __restrict__ X, int rows, int F,
                         float* __restrict__ mean, float* __restrict__ var) {
  __shared__ float s1[256], s2[256];
  int f = blockIdx.x;
  float a = 0.f, b = 0.f;
  for (int i = threadIdx.x; i < rows; i += blockDim.x) {
    float v = X[(size_t)i * F + f];
    a += v; b += v * v;
  }
  s1[threadIdx.x] = a; s2[threadIdx.x] = b;
  __syncthreads();
  for (int s = 128; s > 0; s >>= 1) {
    if (threadIdx.x < s) { s1[threadIdx.x] += s1[threadIdx.x + s]; s2[threadIdx.x] += s2[threadIdx.x + s]; }
    __syncthreads();
  }
  if (threadIdx.x == 0) {
    float m = s1[0] / rows;
    mean[f] = m;
    var[f]  = s2[0] / rows - m * m;              // biased variance (ddof=0)
  }
}

__global__ void bn_apply(const float* __restrict__ X, float* __restrict__ Y,
                         const float* __restrict__ mean, const float* __restrict__ var,
                         const float* __restrict__ g, const float* __restrict__ be,
                         int rows, int F) {
  int i = blockIdx.x * blockDim.x + threadIdx.x;
  if (i < rows * F) {
    int f = i % F;
    Y[i] = g[f] * (X[i] - mean[f]) * rsqrtf(var[f] + 1e-5f) + be[f];
  }
}

__global__ void attn_lr(const float* __restrict__ hp, const float* __restrict__ al,
                        const float* __restrict__ ar, float* __restrict__ el,
                        float* __restrict__ er, int rows) {
  int i = blockIdx.x * blockDim.x + threadIdx.x;
  if (i >= rows * 8) return;
  int v = i >> 3, h = i & 7;
  const float* p = hp + (size_t)v * 1024 + h * 128;
  const float* a = al + h * 128;
  const float* b = ar + h * 128;
  float sl = 0.f, sr = 0.f;
  for (int d = 0; d < 128; ++d) { float x = p[d]; sl += x * a[d]; sr += x * b[d]; }
  el[i] = sl; er[i] = sr;
}

__global__ void attn_e_pre(const float* __restrict__ We, const float* __restrict__ ae,
                           float* __restrict__ c01) {
  int h = threadIdx.x;
  if (h < 8) {
    float s0 = 0.f, s1 = 0.f;
    for (int d = 0; d < 128; ++d) {
      s0 += We[h * 128 + d]        * ae[h * 128 + d];
      s1 += We[1024 + h * 128 + d] * ae[h * 128 + d];
    }
    c01[h] = s0; c01[8 + h] = s1;
  }
}

__global__ void lmax_init(const float* el, const float* er, float* lmax, int n) {
  int i = blockIdx.x * blockDim.x + threadIdx.x;
  if (i < n) lmax[i] = leaky02(el[i] + er[i]);   // self-loop logit (ee == 0)
}

__global__ void edge_logits(const float* __restrict__ el, const float* __restrict__ er,
                            const float* __restrict__ ef, const float* __restrict__ c01,
                            const int* __restrict__ src, const int* __restrict__ dst,
                            float* __restrict__ lg, float* __restrict__ lmax, int e) {
  int i = blockIdx.x * blockDim.x + threadIdx.x;
  if (i >= e * 8) return;
  int ed = i >> 3, h = i & 7;
  int s = src[ed], d = dst[ed];
  float x = el[s * 8 + h] + er[d * 8 + h]
          + ef[ed * 2 + 0] * c01[h] + ef[ed * 2 + 1] * c01[8 + h];
  x = leaky02(x);
  lg[i] = x;
  atomicMaxF(&lmax[d * 8 + h], x);
}

__global__ void denom_init(const float* el, const float* er, const float* lmax,
                           float* den, int n) {
  int i = blockIdx.x * blockDim.x + threadIdx.x;
  if (i < n) den[i] = __expf(leaky02(el[i] + er[i]) - lmax[i]);
}

__global__ void edge_exp(float* __restrict__ lg, const float* __restrict__ lmax,
                         float* __restrict__ den, const int* __restrict__ dst, int e) {
  int i = blockIdx.x * blockDim.x + threadIdx.x;
  if (i >= e * 8) return;
  int ed = i >> 3, h = i & 7, d = dst[ed];
  float z = __expf(lg[i] - lmax[d * 8 + h]);
  lg[i] = z;
  atomicAdd(&den[d * 8 + h], z);
}

__global__ void edge_norm(float* __restrict__ lg, const float* __restrict__ den,
                          const int* __restrict__ dst, int e) {
  int i = blockIdx.x * blockDim.x + threadIdx.x;
  if (i >= e * 8) return;
  int ed = i >> 3, h = i & 7;
  lg[i] /= den[dst[ed] * 8 + h];
}

__global__ void rst_init(const float* __restrict__ hp, const float* __restrict__ el,
                         const float* __restrict__ er, const float* __restrict__ lmax,
                         const float* __restrict__ den, float* __restrict__ rst, int rows) {
  int i = blockIdx.x * blockDim.x + threadIdx.x;
  if (i >= rows * 1024) return;
  int v = i >> 10, h = (i & 1023) >> 7;
  float x = leaky02(el[v * 8 + h] + er[v * 8 + h]);
  float a = __expf(x - lmax[v * 8 + h]) / den[v * 8 + h];
  rst[i] = a * hp[i];
}

__global__ void gat_scatter(const float* __restrict__ hp, const float* __restrict__ a,
                            const int* __restrict__ src, const int* __restrict__ dst,
                            float* __restrict__ rst, int e) {
  int i = blockIdx.x * blockDim.x + threadIdx.x;
  if (i >= e * 256) return;
  int ed = i >> 8, c = (i & 255) << 2, h = c >> 7;
  float w = a[ed * 8 + h];
  const float4 v = *(const float4*)(hp + (size_t)src[ed] * 1024 + c);
  float* q = rst + (size_t)dst[ed] * 1024 + c;
  atomicAdd(q + 0, v.x * w); atomicAdd(q + 1, v.y * w);
  atomicAdd(q + 2, v.z * w); atomicAdd(q + 3, v.w * w);
}

__global__ void add_res(float* __restrict__ rst, const float* __restrict__ hres,
                        const float* __restrict__ bg, int rows) {
  int i = blockIdx.x * blockDim.x + threadIdx.x;
  if (i < rows * 1024) rst[i] += hres[i] + bg[i & 1023];
}

__global__ void head_out(const float* __restrict__ t1, const float* __restrict__ b1,
                         const float* __restrict__ w2, const float* __restrict__ b2,
                         float* __restrict__ out, int rows) {
  int v = blockIdx.x * blockDim.x + threadIdx.x;
  if (v >= rows) return;
  float s = 0.f;
  const float* p = t1 + (size_t)v * 128;
  for (int j = 0; j < 128; ++j) {
    float u = p[j] + b1[j];
    s += (u > 0.f ? u : 0.f) * w2[j];
  }
  out[v] = s + b2[0];
}

// ---------------------------------------------------------------------------
static inline int cdiv(long long a, int b) { return (int)((a + b - 1) / b); }

extern "C" void kernel_launch(void* const* d_in, const int* in_sizes, int n_in,
                              void* d_out, int out_size, void* d_ws, size_t ws_size,
                              hipStream_t stream) {
  (void)in_sizes; (void)n_in; (void)out_size; (void)ws_size;
  const float* x     = (const float*)d_in[0];
  const float* efeat = (const float*)d_in[1];
  const int*   src   = (const int*)d_in[2];
  const int*   dst   = (const int*)d_in[3];
  const float* W1  = (const float*)d_in[4],  *b1  = (const float*)d_in[5];
  const float* g1  = (const float*)d_in[6],  *be1 = (const float*)d_in[7];
  const float* W2  = (const float*)d_in[8],  *b2  = (const float*)d_in[9];
  const float* g2  = (const float*)d_in[10], *be2 = (const float*)d_in[11];
  const float* W3  = (const float*)d_in[12], *b3  = (const float*)d_in[13];
  const float* g3  = (const float*)d_in[14], *be3 = (const float*)d_in[15];
  const float* Wg  = (const float*)d_in[16], *We  = (const float*)d_in[17];
  const float* atl = (const float*)d_in[18], *atr = (const float*)d_in[19];
  const float* ate = (const float*)d_in[20];
  const float* Wres = (const float*)d_in[21], *bg  = (const float*)d_in[22];
  const float* g5  = (const float*)d_in[23], *be5 = (const float*)d_in[24];
  const float* Wa1 = (const float*)d_in[25], *ba1 = (const float*)d_in[26];
  const float* Wa2 = (const float*)d_in[27], *ba2 = (const float*)d_in[28];
  const float* Wc1 = (const float*)d_in[29], *bc1 = (const float*)d_in[30];
  const float* Wc2 = (const float*)d_in[31], *bc2 = (const float*)d_in[32];
  float* out = (float*)d_out;

  // workspace carve-out (256B aligned)
  size_t off = 0;
  auto carve = [&](size_t bytes) {
    void* p = (char*)d_ws + off;
    off += (bytes + 255) & ~(size_t)255;
    return p;
  };
  float* bufA = (float*)carve((size_t)NNODES * 1024 * 4);
  float* bufB = (float*)carve((size_t)NNODES * 1024 * 4);
  float* bufC = (float*)carve((size_t)NNODES * 1024 * 4);
  float* bufD = (float*)carve((size_t)NNODES * 1024 * 4);
  unsigned short* bufBF = (unsigned short*)carve((size_t)NNODES * 1024 * 2);
  unsigned short* bufW  = (unsigned short*)carve((size_t)512 * 1024 * 2);
  float* odeg = (float*)carve(NNODES * 4);
  float* ideg = (float*)carve(NNODES * 4);
  float* snorm = (float*)carve(NNODES * 4);
  float* dnorm = (float*)carve(NNODES * 4);
  float* mean = (float*)carve(1024 * 4);
  float* var  = (float*)carve(1024 * 4);
  float* el = (float*)carve(NNODES * 8 * 4);
  float* er = (float*)carve(NNODES * 8 * 4);
  float* lmax = (float*)carve(NNODES * 8 * 4);
  float* den  = (float*)carve(NNODES * 8 * 4);
  float* ebuf = (float*)carve((size_t)NEDGES * 8 * 4);
  float* c01  = (float*)carve(16 * 4);

  const int T = 256;
  auto gemm = [&](const unsigned short* A, const unsigned short* Bt, float* C,
                  int M, int N, int K) {
    int tiles = (M / 16) * (N / 64);
    wmma_gemm_bf16<<<cdiv(tiles, 8), 256, 0, stream>>>(A, Bt, C, M, N, K);
  };

  // degrees & symmetric norms (self-loop augmented)
  deg_init<<<cdiv(NNODES, T), T, 0, stream>>>(odeg, ideg, NNODES);
  deg_accum<<<cdiv(NEDGES, T), T, 0, stream>>>(src, dst, odeg, ideg, NEDGES);
  deg_norm<<<cdiv(NNODES, T), T, 0, stream>>>(odeg, ideg, snorm, dnorm, NNODES);

  // --- GraphConv + BN layer ---------------------------------------------
  auto gconv = [&](const float* hin, int din, const float* W, const float* b,
                   const float* g, const float* be, int dout, float* hout) {
    scale_to_bf16<<<cdiv((long long)NNODES * din, T), T, 0, stream>>>(hin, snorm, bufBF, NNODES, din);
    w_to_bf16_t<<<cdiv((long long)din * dout, T), T, 0, stream>>>(W, bufW, din, dout);
    gemm(bufBF, bufW, bufB, NNODES, dout, din);                    // hw
    hipMemcpyAsync(bufC, bufB, (size_t)NNODES * dout * 4,
                   hipMemcpyDeviceToDevice, stream);               // agg = self-loop
    scatter_add4<<<cdiv((long long)NEDGES * (dout / 4), T), T, 0, stream>>>(
        bufB, src, dst, bufC, NEDGES, dout / 4, dout);
    post_relu<<<cdiv((long long)NNODES * dout, T), T, 0, stream>>>(bufC, dnorm, b, NNODES, dout);
    bn_stats<<<dout, 256, 0, stream>>>(bufC, NNODES, dout, mean, var);
    bn_apply<<<cdiv((long long)NNODES * dout, T), T, 0, stream>>>(bufC, hout, mean, var, g, be, NNODES, dout);
  };

  gconv(x,    128, W1, b1, g1, be1, 64,  bufA);
  gconv(bufA, 64,  W2, b2, g2, be2, 256, bufA);
  gconv(bufA, 256, W3, b3, g3, be3, 512, bufA);     // h3 = bufA [N,512]

  // --- EdgeGAT -----------------------------------------------------------
  scale_to_bf16<<<cdiv((long long)NNODES * 512, T), T, 0, stream>>>(bufA, nullptr, bufBF, NNODES, 512);
  w_to_bf16_t<<<cdiv(512 * 1024LL, T), T, 0, stream>>>(Wg, bufW, 512, 1024);
  gemm(bufBF, bufW, bufB, NNODES, 1024, 512);       // hp   = bufB
  w_to_bf16_t<<<cdiv(512 * 1024LL, T), T, 0, stream>>>(Wres, bufW, 512, 1024);
  gemm(bufBF, bufW, bufC, NNODES, 1024, 512);       // hres = bufC

  attn_lr<<<cdiv(NNODES * 8LL, T), T, 0, stream>>>(bufB, atl, atr, el, er, NNODES);
  attn_e_pre<<<1, 64, 0, stream>>>(We, ate, c01);
  lmax_init<<<cdiv(NNODES * 8LL, T), T, 0, stream>>>(el, er, lmax, NNODES * 8);
  edge_logits<<<cdiv((long long)NEDGES * 8, T), T, 0, stream>>>(el, er, efeat, c01, src, dst, ebuf, lmax, NEDGES);
  denom_init<<<cdiv(NNODES * 8LL, T), T, 0, stream>>>(el, er, lmax, den, NNODES * 8);
  edge_exp<<<cdiv((long long)NEDGES * 8, T), T, 0, stream>>>(ebuf, lmax, den, dst, NEDGES);
  edge_norm<<<cdiv((long long)NEDGES * 8, T), T, 0, stream>>>(ebuf, den, dst, NEDGES);
  rst_init<<<cdiv((long long)NNODES * 1024, T), T, 0, stream>>>(bufB, el, er, lmax, den, bufD, NNODES);
  gat_scatter<<<cdiv((long long)NEDGES * 256, T), T, 0, stream>>>(bufB, ebuf, src, dst, bufD, NEDGES);
  add_res<<<cdiv((long long)NNODES * 1024, T), T, 0, stream>>>(bufD, bufC, bg, NNODES);

  bn_stats<<<1024, 256, 0, stream>>>(bufD, NNODES, 1024, mean, var);
  bn_apply<<<cdiv((long long)NNODES * 1024, T), T, 0, stream>>>(bufD, bufA, mean, var, g5, be5, NNODES, 1024); // z

  // --- Heads -------------------------------------------------------------
  scale_to_bf16<<<cdiv((long long)NNODES * 1024, T), T, 0, stream>>>(bufA, nullptr, bufBF, NNODES, 1024);
  w_to_bf16_t<<<cdiv(1024 * 128LL, T), T, 0, stream>>>(Wa1, bufW, 1024, 128);
  gemm(bufBF, bufW, bufB, NNODES, 128, 1024);
  head_out<<<cdiv(NNODES, T), T, 0, stream>>>(bufB, ba1, Wa2, ba2, out, NNODES);
  w_to_bf16_t<<<cdiv(1024 * 128LL, T), T, 0, stream>>>(Wc1, bufW, 1024, 128);
  gemm(bufBF, bufW, bufC, NNODES, 128, 1024);
  head_out<<<cdiv(NNODES, T), T, 0, stream>>>(bufC, bc1, Wc2, bc2, out + NNODES, NNODES);
}
